// TGAT_68676527063771
// MI455X (gfx1250) — matrix operand order repaired and checked by reading, not compiled, not run on hardware
//
#include <hip/hip_runtime.h>
#include <hip/hip_bf16.h>
#include <stdint.h>

typedef __attribute__((ext_vector_type(16))) _Float16 v16h;
typedef __attribute__((ext_vector_type(8)))  float    v8f;

#define WMMA(a,b,c) __builtin_amdgcn_wmma_f32_16x16x32_f16(false,(a),false,(b),(short)0,(c),false,false)

__device__ __forceinline__ unsigned f2o(float f){
  unsigned u = __float_as_uint(f);
  return (u & 0x80000000u) ? ~u : (u | 0x80000000u);
}
__device__ __forceinline__ float o2f(unsigned u){
  return (u & 0x80000000u) ? __uint_as_float(u & 0x7fffffffu) : __uint_as_float(~u);
}
__device__ __forceinline__ float sigm(float x){ return 1.f / (1.f + __expf(-x)); }

// ---------------- fragment helpers (ISA 7.12.2 layouts, wave32) ----------------
// A 16x32 f16: lane L, vgpr v -> row m=L%16, k=(v<4?0:16)+8*(L/16)+2*(v&3)
// Per lane that is two contiguous 8-float runs at kb and kb+16 (kb = k0+8*(L/16)).
__device__ __forceinline__ v16h cvt_runs(float4 q0, float4 q1, float4 q2, float4 q3){
  v16h a;
  a[0]=(_Float16)q0.x;  a[1]=(_Float16)q0.y;  a[2]=(_Float16)q0.z;  a[3]=(_Float16)q0.w;
  a[4]=(_Float16)q1.x;  a[5]=(_Float16)q1.y;  a[6]=(_Float16)q1.z;  a[7]=(_Float16)q1.w;
  a[8]=(_Float16)q2.x;  a[9]=(_Float16)q2.y;  a[10]=(_Float16)q2.z; a[11]=(_Float16)q2.w;
  a[12]=(_Float16)q3.x; a[13]=(_Float16)q3.y; a[14]=(_Float16)q3.z; a[15]=(_Float16)q3.w;
  return a;
}
// full: wave-uniform flag, true when row tile fully in-bounds AND k-slice complete
__device__ __forceinline__ v16h a_frag_global(const float* A, long long lda, int row0, int M,
                                              int k0, int K, int lane, bool full){
  int m = row0 + (lane & 15);
  int kb = k0 + 8 * (lane >> 4);
  if (full){
    const float* p = A + (long long)m * lda + kb;   // 32B aligned for all callers
    float4 q0 = ((const float4*)p)[0];
    float4 q1 = ((const float4*)p)[1];
    float4 q2 = ((const float4*)(p + 16))[0];
    float4 q3 = ((const float4*)(p + 16))[1];
    return cvt_runs(q0, q1, q2, q3);
  }
  v16h a;
  bool mv = (m < M);
  #pragma unroll
  for (int v = 0; v < 8; ++v){
    int k = kb + ((v < 4) ? 0 : 16) + 2 * (v & 3);
    float f0 = (mv && k     < K) ? A[(long long)m * lda + k]     : 0.f;
    float f1 = (mv && (k+1) < K) ? A[(long long)m * lda + k + 1] : 0.f;
    a[2*v] = (_Float16)f0; a[2*v+1] = (_Float16)f1;
  }
  return a;
}
// LDS variant: callers guarantee full tiles (H,I multiples of 32), ldh*4 multiple of 16B
__device__ __forceinline__ v16h a_frag_lds(const float* h, int ldh, int k0, int lane){
  int m = lane & 15;
  int kb = k0 + 8 * (lane >> 4);
  const float* p = h + m * ldh + kb;
  float4 q0 = ((const float4*)p)[0];
  float4 q1 = ((const float4*)p)[1];
  float4 q2 = ((const float4*)(p + 16))[0];
  float4 q3 = ((const float4*)(p + 16))[1];
  return cvt_runs(q0, q1, q2, q3);
}
// B fragments pre-packed: one contiguous 32B v16h per (ktile,ntile,lane)
__device__ __forceinline__ v16h b_frag(const _Float16* P, int Ntiles, int kt, int nt, int lane){
  return ((const v16h*)P)[(size_t)(kt * Ntiles + nt) * 32 + lane];
}

// ---------------- utility kernels ----------------
__global__ void fill_kernel(unsigned* p, unsigned v, long long n){
  long long i = (long long)blockIdx.x * blockDim.x + threadIdx.x;
  long long st = (long long)gridDim.x * blockDim.x;
  for (; i < n; i += st) p[i] = v;
}

// Pack B[k][n] (= W[k][n] or W^T) into WMMA B-fragment order:
// B 32x16 f16: lane L, vgpr v -> col n=L%16, k=(v<4?0:16)+8*(L/16)+2*(v&3)
__global__ void pack_b_kernel(const float* __restrict__ W, _Float16* __restrict__ P,
                              int K, int N, int ldw, int transpose){
  int Ntiles = (N + 15) >> 4;
  int Ktiles = (K + 31) >> 5;
  int idx = blockIdx.x * blockDim.x + threadIdx.x;
  int total = Ktiles * Ntiles * 32;
  if (idx >= total) return;
  int lane = idx & 31;
  int tile = idx >> 5;
  int nt = tile % Ntiles;
  int kt = tile / Ntiles;
  int n = nt * 16 + (lane & 15);
  v16h frag;
  #pragma unroll
  for (int v = 0; v < 8; ++v){
    int k = kt * 32 + 8 * (lane >> 4) + ((v < 4) ? 0 : 16) + 2 * (v & 3);
    float f0 = 0.f, f1 = 0.f;
    if (n < N){
      if (k     < K) f0 = transpose ? W[(size_t)n * ldw + k]     : W[(size_t)k * ldw + n];
      if ((k+1) < K) f1 = transpose ? W[(size_t)n * ldw + k + 1] : W[(size_t)(k + 1) * ldw + n];
    }
    frag[2*v] = (_Float16)f0; frag[2*v+1] = (_Float16)f1;
  }
  ((v16h*)P)[(size_t)tile * 32 + lane] = frag;
}

// ---------------- generic WMMA GEMM: Out[M,N] = A[M,K] @ Bpacked + bias ----------------
// one wave/block; block = 16 rows x up to 4 n-tiles (64 cols)
__global__ void gemm_kernel(const float* __restrict__ A, long long lda,
                            const _Float16* __restrict__ P, const float* __restrict__ bias,
                            float* __restrict__ Out, long long ldo,
                            int Mrows, int Ncols, int Kdim){
  int lane = threadIdx.x;
  int row0 = blockIdx.x * 16;
  int Ntiles = (Ncols + 15) >> 4;
  int Ktiles = (Kdim + 31) >> 5;
  int nt0 = blockIdx.y * 4;
  int ntl = Ntiles - nt0; if (ntl > 4) ntl = 4;
  bool fullM = (row0 + 16 <= Mrows);
  v8f acc0 = {}, acc1 = {}, acc2 = {}, acc3 = {};
  for (int kt = 0; kt < Ktiles; ++kt){
    bool full = fullM && (kt * 32 + 32 <= Kdim);
    v16h a = a_frag_global(A, lda, row0, Mrows, kt * 32, Kdim, lane, full);
    v16h b = b_frag(P, Ntiles, kt, nt0 + 0, lane);
    acc0 = WMMA(a, b, acc0);
    if (ntl > 1){ b = b_frag(P, Ntiles, kt, nt0 + 1, lane); acc1 = WMMA(a, b, acc1); }
    if (ntl > 2){ b = b_frag(P, Ntiles, kt, nt0 + 2, lane); acc2 = WMMA(a, b, acc2); }
    if (ntl > 3){ b = b_frag(P, Ntiles, kt, nt0 + 3, lane); acc3 = WMMA(a, b, acc3); }
  }
  int nl = lane & 15, mb = (lane >> 4) * 8;
  for (int j = 0; j < ntl; ++j){
    v8f acc = (j == 0) ? acc0 : (j == 1) ? acc1 : (j == 2) ? acc2 : acc3;
    int n = (nt0 + j) * 16 + nl;
    if (n < Ncols){
      float bv = bias ? bias[n] : 0.f;
      #pragma unroll
      for (int v = 0; v < 8; ++v){
        int m = row0 + mb + v;
        if (m < Mrows) Out[(size_t)m * ldo + n] = acc[v] + bv;
      }
    }
  }
}

// ---------------- GAT kernels ----------------
__global__ void deg_sum_kernel(const int* __restrict__ src, const int* __restrict__ dst,
                               const float* __restrict__ ea, float* __restrict__ sums, int E){
  int e = blockIdx.x * blockDim.x + threadIdx.x;
  if (e >= E) return;
  int d = dst[e];
  atomicAdd(&sums[(size_t)d * 4 + 0], 1.0f);
  #pragma unroll
  for (int j = 0; j < 3; ++j) atomicAdd(&sums[(size_t)d * 4 + 1 + j], ea[(size_t)e * 3 + j]);
}
__global__ void loop_attr_kernel(const float* __restrict__ sums, float* __restrict__ la, int N){
  int n = blockIdx.x * blockDim.x + threadIdx.x;
  if (n >= N) return;
  float deg = fmaxf(sums[(size_t)n * 4], 1.0f);
  #pragma unroll
  for (int j = 0; j < 3; ++j) la[(size_t)n * 3 + j] = sums[(size_t)n * 4 + 1 + j] / deg;
}

__global__ void gat_logit_kernel(const float* __restrict__ xl, const float* __restrict__ xr,
                                 const float* __restrict__ We, const float* __restrict__ att,
                                 const int* __restrict__ src, const int* __restrict__ dst,
                                 const float* __restrict__ ea, const float* __restrict__ la,
                                 float* __restrict__ logit, unsigned* __restrict__ lmax,
                                 int E, int N, int H, int C){
  long long idx = (long long)blockIdx.x * blockDim.x + threadIdx.x;
  long long Ea = (long long)(E + N) * H;
  if (idx >= Ea) return;
  int e = (int)(idx / H), h = (int)(idx % H);
  int s, d; const float* a3;
  if (e < E){ s = src[e]; d = dst[e]; a3 = &ea[(size_t)e * 3]; }
  else      { s = d = e - E;          a3 = &la[(size_t)(e - E) * 3]; }
  float e0 = a3[0], e1 = a3[1], e2 = a3[2];
  int HC = H * C;
  const float* xls = &xl[(size_t)s * HC + h * C];
  const float* xrd = &xr[(size_t)d * HC + h * C];
  const float* ath = &att[h * C];
  const float* W0 = &We[h * C], *W1 = &We[HC + h * C], *W2 = &We[2 * HC + h * C];
  float lg = 0.f;
  for (int c = 0; c < C; ++c){
    float m = xls[c] + xrd[c] + e0 * W0[c] + e1 * W1[c] + e2 * W2[c];
    m = (m > 0.f) ? m : 0.2f * m;          // leaky_relu(0.2)
    lg += ath[c] * m;
  }
  logit[idx] = lg;
  atomicMax(&lmax[(size_t)d * H + h], f2o(lg));
}

__global__ void gat_w_kernel(const float* __restrict__ logit, const int* __restrict__ dst,
                             const unsigned* __restrict__ lmax, float* __restrict__ wbuf,
                             float* __restrict__ denom, int E, int N, int H){
  long long idx = (long long)blockIdx.x * blockDim.x + threadIdx.x;
  long long Ea = (long long)(E + N) * H;
  if (idx >= Ea) return;
  int e = (int)(idx / H), h = (int)(idx % H);
  int d = (e < E) ? dst[e] : (e - E);
  float w = __expf(logit[idx] - o2f(lmax[(size_t)d * H + h]));
  wbuf[idx] = w;
  atomicAdd(&denom[(size_t)d * H + h], w);
}

__global__ void gat_agg_kernel(const float* __restrict__ wbuf, const float* __restrict__ denom,
                               const float* __restrict__ xl, const int* __restrict__ src,
                               const int* __restrict__ dst, float* __restrict__ agg,
                               int E, int N, int H, int C){
  long long idx = (long long)blockIdx.x * blockDim.x + threadIdx.x;
  long long Ea = (long long)(E + N) * H;
  if (idx >= Ea) return;
  int e = (int)(idx / H), h = (int)(idx % H);
  int s, d;
  if (e < E){ s = src[e]; d = dst[e]; } else { s = d = e - E; }
  float alpha = wbuf[idx] / denom[(size_t)d * H + h];
  const float* xs = &xl[(size_t)s * H * C + h * C];
  float* o = &agg[(size_t)d * H * C + h * C];
  for (int c = 0; c < C; ++c) atomicAdd(&o[c], xs[c] * alpha);
}

__global__ void gat_final_kernel(const float* __restrict__ agg, const float* __restrict__ bias,
                                 float* __restrict__ out, long long ldo, int HC, int N){
  long long idx = (long long)blockIdx.x * blockDim.x + threadIdx.x;
  if (idx >= (long long)N * HC) return;
  int n = (int)(idx / HC), j = (int)(idx % HC);
  float v = agg[idx] + bias[j];
  out[(size_t)n * ldo + j] = (v > 0.f) ? v : (__expf(v) - 1.f);  // elu
}

// ---------------- persistent-recurrence LSTM (one direction) ----------------
// one wave handles 16 nodes across all T steps; gates via WMMA; h/c resident in LDS
__global__ void lstm_kernel(const float* __restrict__ in, int I, int T,
                            const _Float16* __restrict__ PWih, const _Float16* __restrict__ PWhh,
                            const float* __restrict__ bias, float* __restrict__ out,
                            int H, int Nn, int dir){
  __shared__ float s_gates[16 * 512];
  __shared__ float s_h[16 * 128];
  __shared__ float s_c[16 * 128];
  int lane = threadIdx.x;
  int row0 = blockIdx.x * 16;
  int G = 4 * H;
  int Gt = G >> 4;                 // gate n-tiles (32 or 16; multiple of 4)
  int KtI = (I + 31) >> 5;
  int KtH = (H + 31) >> 5;
  bool fullM = (row0 + 16 <= Nn);  // I,H are multiples of 32, so only row bound matters
  long long lda = (long long)T * I;
  for (int i = lane; i < 16 * H; i += 32){ s_h[i] = 0.f; s_c[i] = 0.f; }
  __syncthreads();
  for (int step = 0; step < T; ++step){
    int t = dir ? (T - 1 - step) : step;
    const float* xbase = in + (long long)t * I;
    for (int ntc = 0; ntc < Gt; ntc += 4){
      v8f acc0 = {}, acc1 = {}, acc2 = {}, acc3 = {};
      for (int kt = 0; kt < KtI; ++kt){
        v16h a = a_frag_global(xbase, lda, row0, Nn, kt * 32, I, lane, fullM);
        acc0 = WMMA(a, b_frag(PWih, Gt, kt, ntc + 0, lane), acc0);
        acc1 = WMMA(a, b_frag(PWih, Gt, kt, ntc + 1, lane), acc1);
        acc2 = WMMA(a, b_frag(PWih, Gt, kt, ntc + 2, lane), acc2);
        acc3 = WMMA(a, b_frag(PWih, Gt, kt, ntc + 3, lane), acc3);
      }
      for (int kt = 0; kt < KtH; ++kt){
        v16h a = a_frag_lds(s_h, H, kt * 32, lane);
        acc0 = WMMA(a, b_frag(PWhh, Gt, kt, ntc + 0, lane), acc0);
        acc1 = WMMA(a, b_frag(PWhh, Gt, kt, ntc + 1, lane), acc1);
        acc2 = WMMA(a, b_frag(PWhh, Gt, kt, ntc + 2, lane), acc2);
        acc3 = WMMA(a, b_frag(PWhh, Gt, kt, ntc + 3, lane), acc3);
      }
      int nl = lane & 15, mb = (lane >> 4) * 8;
      for (int j = 0; j < 4; ++j){
        v8f acc = (j == 0) ? acc0 : (j == 1) ? acc1 : (j == 2) ? acc2 : acc3;
        int n = (ntc + j) * 16 + nl;
        float bv = bias[n];
        #pragma unroll
        for (int v = 0; v < 8; ++v) s_gates[(mb + v) * G + n] = acc[v] + bv;
      }
    }
    __syncthreads();
    for (int i = lane; i < 16 * H; i += 32){
      int m = i / H, j = i - m * H;
      float gi = s_gates[m * G + j];
      float gf = s_gates[m * G + H + j];
      float gg = s_gates[m * G + 2 * H + j];
      float go = s_gates[m * G + 3 * H + j];
      float c = sigm(gf) * s_c[i] + sigm(gi) * tanhf(gg);
      float h = sigm(go) * tanhf(c);
      s_c[i] = c; s_h[i] = h;
      int node = row0 + m;
      if (node < Nn) out[((size_t)node * T + t) * (size_t)(2 * H) + (size_t)dir * H + j] = h;
    }
    __syncthreads();
  }
}

// ---------------- attention + FC ----------------
__global__ void att_logit_kernel(const float* __restrict__ Xs, const float* __restrict__ attW,
                                 const float* __restrict__ attb, float* __restrict__ lg,
                                 unsigned* __restrict__ gmax, int NT){
  int idx = blockIdx.x * blockDim.x + threadIdx.x;
  if (idx >= NT) return;
  const float* xp = &Xs[(size_t)idx * 128];
  float s = attb[0];
  for (int j = 0; j < 128; ++j) s += xp[j] * attW[j];
  lg[idx] = s;
  atomicMax(gmax, f2o(s));
}
__global__ void att_sum_kernel(const float* __restrict__ lg, const unsigned* __restrict__ gmax,
                               float* __restrict__ gsum, int NT){
  int idx = blockIdx.x * blockDim.x + threadIdx.x;
  if (idx >= NT) return;
  atomicAdd(gsum, __expf(lg[idx] - o2f(*gmax)));
}
__global__ void final_kernel(const float* __restrict__ Xs, const float* __restrict__ lg,
                             const unsigned* __restrict__ gmax, const float* __restrict__ gsum,
                             const float* __restrict__ fcW, const float* __restrict__ fcb,
                             float* __restrict__ out, int N, int T){
  int idx = blockIdx.x * blockDim.x + threadIdx.x;
  if (idx >= N * 4) return;
  int n = idx >> 2, c = idx & 3;
  float gm = o2f(*gmax), gs = *gsum;
  float acc = 0.f;
  for (int t = 0; t < T; ++t){
    float aw = __expf(lg[n * T + t] - gm) / gs;
    const float* xp = &Xs[((size_t)n * T + t) * 128];
    float dt = 0.f;
    for (int j = 0; j < 128; ++j) dt += xp[j] * fcW[j * 4 + c];
    acc += aw * dt;
  }
  out[idx] = acc + fcb[c];
}

// ---------------- host orchestration ----------------
extern "C" void kernel_launch(void* const* d_in, const int* in_sizes, int n_in,
                              void* d_out, int out_size, void* d_ws, size_t ws_size,
                              hipStream_t stream) {
  (void)n_in; (void)out_size; (void)ws_size;
  const int T = 8, IN = 8;
  const float* x      = (const float*)d_in[0];
  const int*   eidx   = (const int*)d_in[1];
  const float* ea     = (const float*)d_in[2];
  const float* g1_Wl  = (const float*)d_in[3];  const float* g1_bl  = (const float*)d_in[4];
  const float* g1_Wr  = (const float*)d_in[5];  const float* g1_br  = (const float*)d_in[6];
  const float* g1_We  = (const float*)d_in[7];  const float* g1_att = (const float*)d_in[8];
  const float* g1_bias= (const float*)d_in[9];
  const float* g2_Wl  = (const float*)d_in[10]; const float* g2_bl  = (const float*)d_in[11];
  const float* g2_Wr  = (const float*)d_in[12]; const float* g2_br  = (const float*)d_in[13];
  const float* g2_We  = (const float*)d_in[14]; const float* g2_att = (const float*)d_in[15];
  const float* g2_bias= (const float*)d_in[16];
  const float* l1_Wih = (const float*)d_in[17]; const float* l1_Whh = (const float*)d_in[18];
  const float* l1_b   = (const float*)d_in[19];
  const float* l2_Wih = (const float*)d_in[20]; const float* l2_Whh = (const float*)d_in[21];
  const float* l2_b   = (const float*)d_in[22];
  const float* att_W  = (const float*)d_in[23]; const float* att_b  = (const float*)d_in[24];
  const float* fc_W   = (const float*)d_in[25]; const float* fc_b   = (const float*)d_in[26];

  const int N  = in_sizes[0] / (T * IN);
  const int E  = in_sizes[1] / 2;
  const int Ea = E + N;
  const int* src = eidx;
  const int* dst = eidx + E;
  float* out = (float*)d_out;

  char* wsb = (char*)d_ws;
  size_t off = 0;
  auto alloc = [&](size_t bytes)->char*{ char* p = wsb + off; off += (bytes + 63) & ~(size_t)63; return p; };
  auto pkb = [](int K, int Nn)->size_t{ return (size_t)((K + 31) / 32) * ((Nn + 15) / 16) * 512 * 2; };

  _Float16* P_g1Wl = (_Float16*)alloc(pkb(8, 512));
  _Float16* P_g1Wr = (_Float16*)alloc(pkb(8, 512));
  _Float16* P_g2Wl = (_Float16*)alloc(pkb(512, 32));
  _Float16* P_g2Wr = (_Float16*)alloc(pkb(512, 32));
  _Float16 *P_l1Wih[2], *P_l1Whh[2], *P_l2Wih[2], *P_l2Whh[2];
  for (int d = 0; d < 2; ++d){
    P_l1Wih[d] = (_Float16*)alloc(pkb(32, 512));
    P_l1Whh[d] = (_Float16*)alloc(pkb(128, 512));
    P_l2Wih[d] = (_Float16*)alloc(pkb(256, 256));
    P_l2Whh[d] = (_Float16*)alloc(pkb(64, 256));
  }
  float* sums   = (float*)alloc((size_t)N * 4 * 4);
  float* la     = (float*)alloc((size_t)N * 3 * 4);
  float* seq32  = (float*)alloc((size_t)N * T * 32 * 4);
  float* lstm1o = (float*)alloc((size_t)N * T * 256 * 4);
  float* lstm2o = (float*)alloc((size_t)N * T * 128 * 4);
  float* alog   = (float*)alloc((size_t)N * T * 4);
  unsigned* gscal = (unsigned*)alloc(64);       // [0]=gmax (ordered), [1]=gsum (f32)
  float* xl1 = (float*)alloc((size_t)N * 512 * 4);
  float* xr1 = (float*)alloc((size_t)N * 512 * 4);
  float* h1  = (float*)alloc((size_t)N * 512 * 4);
  float* xl2 = (float*)alloc((size_t)N * 32 * 4);
  float* xr2 = (float*)alloc((size_t)N * 32 * 4);
  float* logit1 = (float*)alloc((size_t)Ea * 8 * 4);
  float* w1     = (float*)alloc((size_t)Ea * 8 * 4);
  float* logit2 = (float*)alloc((size_t)Ea * 4);
  float* w2     = (float*)alloc((size_t)Ea * 4);
  // zero-region per timestep: lmax1 | denom1 | agg1 | lmax2 | denom2 | agg2
  long long zcount = (long long)N * (8 + 8 + 512 + 1 + 1 + 32);
  unsigned* zreg  = (unsigned*)alloc((size_t)zcount * 4);
  unsigned* lmax1 = zreg;
  float*    den1  = (float*)(zreg + (size_t)N * 8);
  float*    agg1  = (float*)(zreg + (size_t)N * 16);
  unsigned* lmax2 = zreg + (size_t)N * 16 + (size_t)N * 512;
  float*    den2  = (float*)(lmax2 + N);
  float*    agg2  = (float*)(lmax2 + 2 * (size_t)N);

  auto cdiv = [](long long a, long long b)->int{ return (int)((a + b - 1) / b); };
  auto pack = [&](const float* W, _Float16* P, int K, int Nn, int ldw, int tr){
    int total = ((K + 31) / 32) * ((Nn + 15) / 16) * 32;
    pack_b_kernel<<<cdiv(total, 256), 256, 0, stream>>>(W, P, K, Nn, ldw, tr);
  };

  // weight packing (once)
  pack(g1_Wl, P_g1Wl, 8, 512, 512, 0);
  pack(g1_Wr, P_g1Wr, 8, 512, 512, 0);
  pack(g2_Wl, P_g2Wl, 512, 32, 32, 0);
  pack(g2_Wr, P_g2Wr, 512, 32, 32, 0);
  for (int d = 0; d < 2; ++d){
    pack(l1_Wih + (size_t)d * 512 * 32,  P_l1Wih[d], 32, 512, 32, 1);
    pack(l1_Whh + (size_t)d * 512 * 128, P_l1Whh[d], 128, 512, 128, 1);
    pack(l2_Wih + (size_t)d * 256 * 256, P_l2Wih[d], 256, 256, 256, 1);
    pack(l2_Whh + (size_t)d * 256 * 64,  P_l2Whh[d], 64, 256, 64, 1);
  }
  // self-loop edge attrs (mean of incoming)
  fill_kernel<<<cdiv((long long)N * 4, 256), 256, 0, stream>>>((unsigned*)sums, 0u, (long long)N * 4);
  deg_sum_kernel<<<cdiv(E, 256), 256, 0, stream>>>(src, dst, ea, sums, E);
  loop_attr_kernel<<<cdiv(N, 256), 256, 0, stream>>>(sums, la, N);

  for (int t = 0; t < T; ++t){
    fill_kernel<<<cdiv(zcount, 256), 256, 0, stream>>>(zreg, 0u, zcount);
    // GAT layer 1 (H=8, C=64)
    dim3 gg1(cdiv(N, 16), 8);     // 512 cols -> 32 n-tiles -> 8 y-blocks
    gemm_kernel<<<gg1, 32, 0, stream>>>(x + (size_t)t * N * IN, IN, P_g1Wl, g1_bl, xl1, 512, N, 512, 8);
    gemm_kernel<<<gg1, 32, 0, stream>>>(x + (size_t)t * N * IN, IN, P_g1Wr, g1_br, xr1, 512, N, 512, 8);
    gat_logit_kernel<<<cdiv((long long)Ea * 8, 256), 256, 0, stream>>>(xl1, xr1, g1_We, g1_att, src, dst, ea, la, logit1, lmax1, E, N, 8, 64);
    gat_w_kernel<<<cdiv((long long)Ea * 8, 256), 256, 0, stream>>>(logit1, dst, lmax1, w1, den1, E, N, 8);
    gat_agg_kernel<<<cdiv((long long)Ea * 8, 256), 256, 0, stream>>>(w1, den1, xl1, src, dst, agg1, E, N, 8, 64);
    gat_final_kernel<<<cdiv((long long)N * 512, 256), 256, 0, stream>>>(agg1, g1_bias, h1, 512, 512, N);
    // GAT layer 2 (H=1, C=32)
    dim3 gg2(cdiv(N, 16), 1);
    gemm_kernel<<<gg2, 32, 0, stream>>>(h1, 512, P_g2Wl, g2_bl, xl2, 32, N, 32, 512);
    gemm_kernel<<<gg2, 32, 0, stream>>>(h1, 512, P_g2Wr, g2_br, xr2, 32, N, 32, 512);
    gat_logit_kernel<<<cdiv((long long)Ea, 256), 256, 0, stream>>>(xl2, xr2, g2_We, g2_att, src, dst, ea, la, logit2, lmax2, E, N, 1, 32);
    gat_w_kernel<<<cdiv((long long)Ea, 256), 256, 0, stream>>>(logit2, dst, lmax2, w2, den2, E, N, 1);
    gat_agg_kernel<<<cdiv((long long)Ea, 256), 256, 0, stream>>>(w2, den2, xl2, src, dst, agg2, E, N, 1, 32);
    gat_final_kernel<<<cdiv((long long)N * 32, 256), 256, 0, stream>>>(agg2, g2_bias, seq32 + (size_t)t * 32, (long long)T * 32, 32, N);
  }
  // BiLSTM 1: 32 -> 2x128
  for (int d = 0; d < 2; ++d)
    lstm_kernel<<<cdiv(N, 16), 32, 0, stream>>>(seq32, 32, T, P_l1Wih[d], P_l1Whh[d], l1_b + (size_t)d * 512, lstm1o, 128, N, d);
  // BiLSTM 2: 256 -> 2x64
  for (int d = 0; d < 2; ++d)
    lstm_kernel<<<cdiv(N, 16), 32, 0, stream>>>(lstm1o, 256, T, P_l2Wih[d], P_l2Whh[d], l2_b + (size_t)d * 256, lstm2o, 64, N, d);
  // global-softmax attention + FC
  fill_kernel<<<1, 32, 0, stream>>>(gscal, 0u, 2);
  att_logit_kernel<<<cdiv((long long)N * T, 256), 256, 0, stream>>>(lstm2o, att_W, att_b, alog, gscal, N * T);
  att_sum_kernel<<<cdiv((long long)N * T, 256), 256, 0, stream>>>(alog, gscal, (float*)(gscal + 1), N * T);
  final_kernel<<<cdiv((long long)N * 4, 256), 256, 0, stream>>>(lstm2o, alog, gscal, (const float*)(gscal + 1), fc_W, fc_b, out, N, T);
}
